// AttentionLayer_84172769068274
// MI455X (gfx1250) — matrix operand order repaired and checked by reading, not compiled, not run on hardware
//
#include <hip/hip_runtime.h>
#include <hip/hip_bf16.h>

typedef __attribute__((ext_vector_type(16))) _Float16 v16h;
typedef __attribute__((ext_vector_type(8)))  _Float16 v8h;
typedef __attribute__((ext_vector_type(4)))  _Float16 v4h;
typedef __attribute__((ext_vector_type(8)))  float    v8f;
typedef __attribute__((ext_vector_type(4)))  float    v4f;

// Problem sizes
#define BATCH 512
#define TDIM  200       // T (K of GEMM, and softmax axis)
#define DDIM  512       // D
// Padded tiling
#define KT_TILES 7      // K padded to 7 * 32 = 224
#define NJ    7         // n-tiles per wave (14 n-tiles total, N padded to 224)
#define NP    224
#define DBLK  128       // d-rows per workgroup
#define NTHREADS 256    // 8 waves: 4 row-pairs x 2 column-halves

// LDS layout. Pitches are 232 halves (116 dwords): multiple of 8 halves keeps
// all b128 fragment loads 16B-aligned; 116-dword row stride gives 2-way (vs
// 4-8-way) bank conflicts on the transpose stores.
#define WPITCH 232      // halves per Wt row (n-major, k contiguous)
#define XPITCH 232      // halves per X row  (d-major, t contiguous)
#define SPITCH 228      // floats per S row
#define OFF_WT 0                          // [224][WPITCH]  f16 = 103936 B
#define OFF_XL 103936                     // [DBLK][XPITCH] f16 =  59392 B
#define OFF_SL 163328                     // [DBLK][SPITCH] f32 = 116736 B
#define OFF_BL 280064                     // [NP] f32      =    896 B
#define OFF_RDEN 280960                   // [DBLK][2] f32
#define OFF_RNUM 281984                   // [DBLK][2] f32
#define SHMEM_BYTES 283008

__device__ __forceinline__ v16h ldsA16(const _Float16* p) {
    v8h lo = *(const v8h*)(p);
    v8h hi = *(const v8h*)(p + 16);      // A: second half is +16 halves
    return __builtin_shufflevector(lo, hi, 0,1,2,3,4,5,6,7,8,9,10,11,12,13,14,15);
}
__device__ __forceinline__ v16h ldsB16(const _Float16* p) {
    v8h lo = *(const v8h*)(p);
    v8h hi = *(const v8h*)(p + 16);      // B: second half is +16 halves
    return __builtin_shufflevector(lo, hi, 0,1,2,3,4,5,6,7,8,9,10,11,12,13,14,15);
}

__global__ __launch_bounds__(NTHREADS) void attn_fused_wmma(
    const float* __restrict__ inp,   // [B, T, D]
    const float* __restrict__ Wm,    // [T, T]  (W[t*T + s])
    const float* __restrict__ bias,  // [T]
    float* __restrict__ out)         // [B, D]
{
    extern __shared__ __align__(64) char smem[];
    _Float16* Wt = (_Float16*)(smem + OFF_WT);   // Wt[s][t] = W[t][s], zero-padded
    _Float16* Xl = (_Float16*)(smem + OFF_XL);   // Xl[d][t] = inp[b][t][dblk+d], zero-padded
    float*    Sl = (float*)   (smem + OFF_SL);
    float*    bl = (float*)   (smem + OFF_BL);
    float*    rden = (float*) (smem + OFF_RDEN);
    float*    rnum = (float*) (smem + OFF_RNUM);

    const int tid  = threadIdx.x;
    const int wg   = blockIdx.x;
    const int b    = wg >> 2;                    // 4 d-blocks per batch
    const int dblk = (wg & 3) * DBLK;

    // ---- zero ONLY the padding (disjoint dwords from the fills below) ----
    {
        unsigned* w32 = (unsigned*)(smem + OFF_WT);
        unsigned* x32 = (unsigned*)(smem + OFF_XL);
        // Wt pad rows s = 200..223 (full rows)
        for (int i = tid; i < 24 * 116; i += NTHREADS) w32[200 * 116 + i] = 0u;
        // Wt pad cols k = 200..231 for s < 200 (dwords 100..115 of each row)
        for (int i = tid; i < 200 * 16; i += NTHREADS)
            w32[(i >> 4) * 116 + 100 + (i & 15)] = 0u;
        // Xl pad cols t = 200..231 for d < 128
        for (int i = tid; i < 128 * 16; i += NTHREADS)
            x32[(i >> 4) * 116 + 100 + (i & 15)] = 0u;
    }

    // ---- stage W transposed: Wt[s][t] = W[t][s], 4 t per thread (ds_store_b64) ----
    for (int i = tid; i < (TDIM / 4) * TDIM; i += NTHREADS) {   // 10000 units
        int t4 = i / TDIM;
        int s  = i - t4 * TDIM;                  // consecutive lanes -> coalesced reads
        v4h v;
        #pragma unroll
        for (int q = 0; q < 4; ++q) v[q] = (_Float16)Wm[(4 * t4 + q) * TDIM + s];
        *(v4h*)(Wt + s * WPITCH + 4 * t4) = v;
    }
    // ---- stage X transposed: Xl[d][t], 4 t per thread (ds_store_b64) ----
    for (int i = tid; i < DBLK * (TDIM / 4); i += NTHREADS) {   // 6400 units
        int d  = i & (DBLK - 1);                 // consecutive lanes -> coalesced reads
        int tg = i >> 7;
        const float* g = inp + ((size_t)b * TDIM + 4 * tg) * DDIM + dblk + d;
        v4h v;
        #pragma unroll
        for (int q = 0; q < 4; ++q) v[q] = (_Float16)g[q * DDIM];
        *(v4h*)(Xl + d * XPITCH + 4 * tg) = v;
    }
    if (tid < TDIM) bl[tid] = bias[tid];
    __syncthreads();

    // ---- GEMM: wave owns 2 M-tiles (32 rows) x 7 N-tiles (112 cols) ----
    const int wave  = tid >> 5;
    const int lane  = tid & 31;
    const int mpair = wave >> 1;                 // 0..3
    const int nhalf = wave & 1;                  // 0..1
    const int m0    = mpair * 32;
    const int lrow  = lane & 15;                 // row (A) / column (B,C) in tile
    const int lhi   = lane >> 4;                 // half-wave selector
    const int ncol0 = nhalf * (NJ * 16) + lrow;  // first owned column for this lane

    v8f acc[2 * NJ];
    #pragma unroll
    for (int j = 0; j < 2 * NJ; ++j) { v8f z = {}; acc[j] = z; }

    for (int kt = 0; kt < KT_TILES; ++kt) {
        const int k0 = kt * 32;
        const _Float16* pa = &Xl[(m0 + lrow) * XPITCH + k0 + lhi * 8];
        v16h a0 = ldsA16(pa);                    // rows m0..m0+15
        v16h a1 = ldsA16(pa + 16 * XPITCH);      // rows m0+16..m0+31

        #pragma unroll
        for (int j = 0; j < NJ; ++j) {
            const _Float16* pb = &Wt[(ncol0 + j * 16) * WPITCH + k0 + lhi * 8];
            v16h bf = ldsB16(pb);
            acc[2*j]   = __builtin_amdgcn_wmma_f32_16x16x32_f16(
                false, a0, false, bf, (short)0, acc[2*j],   false, false);
            acc[2*j+1] = __builtin_amdgcn_wmma_f32_16x16x32_f16(
                false, a1, false, bf, (short)0, acc[2*j+1], false, false);
        }
    }

    // ---- spill scores: C layout lane=col, VGPR r -> row r + 8*lhi ----
    #pragma unroll
    for (int j = 0; j < NJ; ++j) {
        const int n = ncol0 + j * 16;
        #pragma unroll
        for (int r = 0; r < 8; ++r) {
            Sl[(m0      + 8 * lhi + r) * SPITCH + n] = acc[2*j][r];
            Sl[(m0 + 16 + 8 * lhi + r) * SPITCH + n] = acc[2*j+1][r];
        }
    }
    __syncthreads();

    // ---- fused bias + tanh + softmax + dot, single pass, 2 threads per row ----
    // tanh output is in [-1,1], so exp() needs no max-shift (identical math to
    // the reference's shifted softmax, and numerically safe).
    const int row  = tid & (DBLK - 1);
    const int half = tid >> 7;                   // 0 or 1
    const int nbeg = half * (TDIM / 2);          // 0 or 100 (multiple of 4)
    const float*    srow = &Sl[row * SPITCH + nbeg];
    const float*    blp  = &bl[nbeg];
    const _Float16* xrow = &Xl[row * XPITCH + nbeg];

    float den = 0.0f, num = 0.0f;
    for (int n = 0; n < TDIM / 2; n += 4) {
        v4f s4 = *(const v4f*)(srow + n);        // 16B aligned
        v4f b4 = *(const v4f*)(blp + n);
        v4h x4 = *(const v4h*)(xrow + n);        // 8B aligned
        #pragma unroll
        for (int q = 0; q < 4; ++q) {
            float s  = s4[q] + b4[q];
            float e2 = __expf(2.0f * s);         // tanh(s) = 1 - 2/(e^{2s}+1)
            float v  = 1.0f - 2.0f / (e2 + 1.0f);
            float e  = __expf(v);                // v in [-1,1]: always safe
            den += e;
            num += e * (float)x4[q];
        }
    }
    rden[row * 2 + half] = den;
    rnum[row * 2 + half] = num;
    __syncthreads();

    if (half == 0) {
        float d    = rden[row * 2] + rden[row * 2 + 1];
        float nsum = rnum[row * 2] + rnum[row * 2 + 1];
        out[(size_t)b * DDIM + dblk + row] = nsum / d;
    }
}

extern "C" void kernel_launch(void* const* d_in, const int* in_sizes, int n_in,
                              void* d_out, int out_size, void* d_ws, size_t ws_size,
                              hipStream_t stream) {
    const float* inp  = (const float*)d_in[0];   // [512, 200, 512] f32
    const float* Wm   = (const float*)d_in[1];   // [200, 200] f32
    const float* bias = (const float*)d_in[2];   // [200] f32
    float* out = (float*)d_out;                  // [512, 512] f32

    dim3 grid(BATCH * (DDIM / DBLK));            // 2048 workgroups
    dim3 block(NTHREADS);
    attn_fused_wmma<<<grid, block, SHMEM_BYTES, stream>>>(inp, Wm, bias, out);
}
